// EncryptedCrossAttention_83184926589619
// MI455X (gfx1250) — matrix operand order repaired and checked by reading, not compiled, not run on hardware
//
#include <hip/hip_runtime.h>

typedef __attribute__((ext_vector_type(16))) __bf16 v16bf;
typedef __attribute__((ext_vector_type(8)))  float  v8f;

#define H_HEADS 8
#define DHEAD   64
#define SEQ     2048
#define INNER   512    // H_HEADS * DHEAD
#define DMODEL  1024
#define NBATCH  4

union FragU { v16bf v; unsigned short u[16]; };
union U4 { uint4 q; unsigned short us[8]; };

__device__ __forceinline__ unsigned short f32_to_bf16(float f) {
  unsigned int x = __float_as_uint(f);
  unsigned int r = (x + 0x7FFFu + ((x >> 16) & 1u)) >> 16;  // round-to-nearest-even
  return (unsigned short)r;
}

__device__ __forceinline__ v8f vzero8() {
  v8f z;
#pragma unroll
  for (int i = 0; i < 8; ++i) z[i] = 0.0f;
  return z;
}

__device__ __forceinline__ v8f wmma_bf16(v16bf a, v16bf b, v8f c) {
  // D = A(16x32 bf16) * B(32x16 bf16) + C(16x16 f32)
  return __builtin_amdgcn_wmma_f32_16x16x32_bf16(false, a, false, b, (short)0, c,
                                                 false, false);
}

// ---- fragment gathers from LDS (layouts per CDNA5 ISA 7.12.2, wave32) ----

// A fragment: 16x32 bf16 tile at (row0, col0) in a row-major LDS buffer.
// lane 0-15: M=lane, K in {0..7,16..23}; lane 16-31: M=lane-16, K in {8..15,24..31}
// Per lane: two contiguous 8-short runs -> 2x ds_load_b128.
__device__ __forceinline__ v16bf load_frag_a(const unsigned short* s, int stride,
                                             int row0, int col0, int lane) {
  const int hlf = lane >> 4;
  const int m = lane & 15;
  FragU f;
  const unsigned short* rp = s + (row0 + m) * stride + col0 + hlf * 8;
#pragma unroll
  for (int i = 0; i < 8; ++i) f.u[i] = rp[i];          // K base+0..7
#pragma unroll
  for (int i = 0; i < 8; ++i) f.u[8 + i] = rp[16 + i]; // K base+16..23
  return f.v;
}

// B fragment where LDS holds B transposed, i.e. memory is [N x K] row-major:
// B[k][n] = s[(n0+n)*stride + k0 + k].
// lane 0-15: N=lane, K in {0..7,16..23}; lane 16-31: N=lane-16, K in {8..15,24..31}
// Per lane: two contiguous 8-short runs -> 2x ds_load_b128.
__device__ __forceinline__ v16bf load_frag_b_nk(const unsigned short* s, int stride,
                                                int n0, int k0, int lane) {
  const int hlf = lane >> 4;
  const int n = lane & 15;
  FragU f;
  const unsigned short* rp = s + (n0 + n) * stride + k0 + hlf * 8;
#pragma unroll
  for (int i = 0; i < 8; ++i) f.u[i] = rp[i];
#pragma unroll
  for (int i = 0; i < 8; ++i) f.u[8 + i] = rp[16 + i];
  return f.v;
}

// =====================================================================
// GEMM + bias: C[M x N] = A[M x K] * B[K x N] + bias, bf16 WMMA, f32 acc.
// BM=128, BN=64, BK=32. 256 threads = 8 waves in 4(M) x 2(N); each wave
// computes a 32x32 tile (4 WMMA accumulators). B is staged transposed in
// LDS so B fragments are contiguous b128 gathers.
// OUT_MODE 0: bf16 output, head-split [B][H][SEQ][64] (for Q/K/V, scale folded)
// OUT_MODE 1: f32 output, plain [M x N] (final projection)
// =====================================================================
template <typename AT, int OUT_MODE>
__global__ __launch_bounds__(256) void proj_kernel(const AT* __restrict__ A,
                                                   const float* __restrict__ Bw,
                                                   const float* __restrict__ bias,
                                                   void* __restrict__ Out,
                                                   int Kdim, int N, float out_scale) {
  __shared__ __align__(16) unsigned short sA[128 * 40];  // [M x K], stride 40
  __shared__ __align__(16) unsigned short sBt[64 * 40];  // [N x K] (transposed), stride 40

  const int tid = threadIdx.x;
  const int lane = tid & 31;
  const int wid = tid >> 5;
  const int wm = wid >> 1;
  const int wn = wid & 1;
  const int hlf = lane >> 4;
  const int ln = lane & 15;

  const int n0 = blockIdx.x * 64;
  const int m0 = blockIdx.y * 128;

  v8f acc[2][2];
#pragma unroll
  for (int i = 0; i < 2; ++i)
#pragma unroll
    for (int j = 0; j < 2; ++j) acc[i][j] = vzero8();

  for (int k0 = 0; k0 < Kdim; k0 += 32) {
    __syncthreads();
    // prefetch next tiles into cache (global_prefetch_b8)
    if (k0 + 32 < Kdim) {
      int r = tid & 127, hb = tid >> 7;
      __builtin_prefetch((const char*)(A + (size_t)(m0 + r) * Kdim + (k0 + 32)) + hb * 64, 0, 3);
      int rb = tid >> 3, cb = tid & 7;
      __builtin_prefetch((const char*)(Bw + (size_t)(k0 + 32 + rb) * N + n0) + cb * 32, 0, 3);
    }
    // stage A tile 128x32 -> bf16 LDS (row-major)
    if constexpr (sizeof(AT) == 4) {
      const float4* gA = (const float4*)A;
      const int rs = Kdim >> 2;
      for (int i = tid; i < 128 * 8; i += 256) {
        int r = i >> 3, c = i & 7;
        float4 fv = gA[(size_t)(m0 + r) * rs + (k0 >> 2) + c];
        unsigned short* dp = &sA[r * 40 + c * 4];
        dp[0] = f32_to_bf16(fv.x);
        dp[1] = f32_to_bf16(fv.y);
        dp[2] = f32_to_bf16(fv.z);
        dp[3] = f32_to_bf16(fv.w);
      }
    } else {  // bf16 input (attention output)
      const uint4* gA = (const uint4*)A;
      const int rs = Kdim >> 3;
      for (int i = tid; i < 128 * 4; i += 256) {
        int r = i >> 2, c = i & 3;
        *(uint4*)&sA[r * 40 + c * 8] = gA[(size_t)(m0 + r) * rs + (k0 >> 3) + c];
      }
    }
    // stage B tile 32x64 -> bf16 LDS transposed [n][k]
    {
      const float4* gB = (const float4*)Bw;
      const int rs = N >> 2;
      for (int i = tid; i < 32 * 16; i += 256) {
        int r = i >> 4, c = i & 15;  // r = k row, c = float4 along n
        float4 fv = gB[(size_t)(k0 + r) * rs + (n0 >> 2) + c];
        int nb = c * 4;
        sBt[(nb + 0) * 40 + r] = f32_to_bf16(fv.x);
        sBt[(nb + 1) * 40 + r] = f32_to_bf16(fv.y);
        sBt[(nb + 2) * 40 + r] = f32_to_bf16(fv.z);
        sBt[(nb + 3) * 40 + r] = f32_to_bf16(fv.w);
      }
    }
    __syncthreads();

    v16bf af[2], bf[2];
#pragma unroll
    for (int mt = 0; mt < 2; ++mt)
      af[mt] = load_frag_a(sA, 40, wm * 32 + mt * 16, 0, lane);
#pragma unroll
    for (int nt = 0; nt < 2; ++nt)
      bf[nt] = load_frag_b_nk(sBt, 40, wn * 32 + nt * 16, 0, lane);
#pragma unroll
    for (int mt = 0; mt < 2; ++mt)
#pragma unroll
      for (int nt = 0; nt < 2; ++nt)
        acc[mt][nt] = wmma_bf16(af[mt], bf[nt], acc[mt][nt]);
  }

  // epilogue: bias + scale + store
#pragma unroll
  for (int mt = 0; mt < 2; ++mt)
#pragma unroll
    for (int nt = 0; nt < 2; ++nt) {
#pragma unroll
      for (int v = 0; v < 8; ++v) {
        int gm = m0 + wm * 32 + mt * 16 + v + 8 * hlf;
        int gn = n0 + wn * 32 + nt * 16 + ln;
        float val = (acc[mt][nt][v] + bias[gn]) * out_scale;
        if constexpr (OUT_MODE == 0) {
          int b = gm >> 11, ns = gm & 2047;   // SEQ == 2048
          int h = gn >> 6, d = gn & 63;
          ((unsigned short*)Out)[((((size_t)b * H_HEADS + h) * SEQ) + ns) * DHEAD + d] =
              f32_to_bf16(val);
        } else {
          ((float*)Out)[(size_t)gm * N + gn] = val;
        }
      }
    }
}

// =====================================================================
// Fused flash attention. Grid: (SEQ/128, H, B); 128 threads = 4 waves,
// each wave owns 32 query rows (2 m-tiles). KV processed in 64-row tiles.
// Online softmax: row-max via lane-group shuffle butterfly; row-SUM via an
// extra WMMA against an all-ones B matrix (l is just a 5th accumulator in
// C layout, no cross-lane reduction needed). 36 WMMAs per wave per KV tile.
// =====================================================================
__global__ __launch_bounds__(128) void attn_kernel(const unsigned short* __restrict__ Qg,
                                                   const unsigned short* __restrict__ Kg,
                                                   const unsigned short* __restrict__ Vg,
                                                   unsigned short* __restrict__ Og) {
  __shared__ __align__(16) unsigned short sQ[128 * 72];   // [q][d]
  __shared__ __align__(16) unsigned short sK[64 * 72];    // [j][d]  (= B^T for S)
  __shared__ __align__(16) unsigned short sVt[64 * 72];   // [d][j]  (= B^T for PV)
  __shared__ __align__(16) unsigned short sP[8 * 16 * 72];  // per (wave,mtile) P staging

  const int tid = threadIdx.x;
  const int lane = tid & 31;
  const int wid = tid >> 5;
  const int hlf = lane >> 4;
  const int ln = lane & 15;

  const int b = blockIdx.z, h = blockIdx.y;
  const int q0 = blockIdx.x * 128;

  const size_t headO = ((size_t)(b * H_HEADS + h)) * SEQ * DHEAD;

  // constant all-ones B fragment (for rowsum-by-WMMA)
  FragU onesu;
#pragma unroll
  for (int i = 0; i < 16; ++i) onesu.u[i] = 0x3F80;  // bf16 1.0
  const v16bf onesB = onesu.v;

  // stage Q tile (128 x 64 bf16) once
  {
    const uint4* gq = (const uint4*)(Qg + headO + (size_t)q0 * DHEAD);
    for (int i = tid; i < 128 * 8; i += 128) {
      int r = i >> 3, c = i & 7;
      *(uint4*)&sQ[r * 72 + c * 8] = gq[(size_t)r * 8 + c];
    }
  }
  __syncthreads();

  // per-wave Q A-fragments: 2 m-tiles x 2 k-steps
  v16bf qfrag[2][2];
#pragma unroll
  for (int mt = 0; mt < 2; ++mt)
#pragma unroll
    for (int ks = 0; ks < 2; ++ks)
      qfrag[mt][ks] = load_frag_a(sQ, 72, wid * 32 + mt * 16, ks * 32, lane);

  float mstate[2][8];
  v8f accO[2][4];
  v8f accL[2];  // running row-sum l (all 16 columns identical)
#pragma unroll
  for (int mt = 0; mt < 2; ++mt) {
#pragma unroll
    for (int v = 0; v < 8; ++v) mstate[mt][v] = -1e30f;
#pragma unroll
    for (int t = 0; t < 4; ++t) accO[mt][t] = vzero8();
    accL[mt] = vzero8();
  }

  for (int kv0 = 0; kv0 < SEQ; kv0 += 64) {
    __syncthreads();  // previous tile fully consumed
    // prefetch next KV tile (global_prefetch_b8): 128 thr x 64B covers 8KB each
    if (kv0 + 64 < SEQ) {
      const unsigned char* nk =
          (const unsigned char*)(Kg + headO + (size_t)(kv0 + 64) * DHEAD);
      const unsigned char* nv =
          (const unsigned char*)(Vg + headO + (size_t)(kv0 + 64) * DHEAD);
      __builtin_prefetch(nk + tid * 64, 0, 3);
      __builtin_prefetch(nv + tid * 64, 0, 3);
    }
    const uint4* gk = (const uint4*)(Kg + headO + (size_t)kv0 * DHEAD);
    const uint4* gv = (const uint4*)(Vg + headO + (size_t)kv0 * DHEAD);
    for (int i = tid; i < 64 * 8; i += 128) {
      int r = i >> 3, c = i & 7;
      *(uint4*)&sK[r * 72 + c * 8] = gk[(size_t)r * 8 + c];  // [j][d]
      U4 u;                                                  // V transposed -> [d][j]
      u.q = gv[(size_t)r * 8 + c];
      int d0 = c * 8;
#pragma unroll
      for (int e = 0; e < 8; ++e) sVt[(d0 + e) * 72 + r] = u.us[e];
    }
    __syncthreads();

#pragma unroll
    for (int mt = 0; mt < 2; ++mt) {
      // S(16x64) = Q(16x64) * K^T : 4 n-tiles, 2 k-steps each
      v8f s[4];
#pragma unroll
      for (int nt = 0; nt < 4; ++nt) {
        v8f c = vzero8();
        c = wmma_bf16(qfrag[mt][0], load_frag_b_nk(sK, 72, nt * 16, 0, lane), c);
        c = wmma_bf16(qfrag[mt][1], load_frag_b_nk(sK, 72, nt * 16, 32, lane), c);
        s[nt] = c;
      }

      // row-max via shuffle butterfly within the row's 16-lane group
      float pscale[8];
#pragma unroll
      for (int v = 0; v < 8; ++v) {
        float mx = fmaxf(fmaxf(s[0][v], s[1][v]), fmaxf(s[2][v], s[3][v]));
#pragma unroll
        for (int msk = 1; msk < 16; msk <<= 1) mx = fmaxf(mx, __shfl_xor(mx, msk, 32));
        float mnew = fmaxf(mstate[mt][v], mx);
        pscale[v] = __expf(mstate[mt][v] - mnew);
        mstate[mt][v] = mnew;
#pragma unroll
        for (int nt = 0; nt < 4; ++nt) s[nt][v] = __expf(s[nt][v] - mnew);
      }
      // rescale running accumulators (O and l)
#pragma unroll
      for (int nt = 0; nt < 4; ++nt)
#pragma unroll
        for (int v = 0; v < 8; ++v) accO[mt][nt][v] *= pscale[v];
#pragma unroll
      for (int v = 0; v < 8; ++v) accL[mt][v] *= pscale[v];

      // stage P (16 x 64 bf16) in (wave,mtile)-private LDS, rebuild as A frags
      unsigned short* pw = &sP[(wid * 2 + mt) * 16 * 72];
#pragma unroll
      for (int nt = 0; nt < 4; ++nt)
#pragma unroll
        for (int v = 0; v < 8; ++v)
          pw[(v + 8 * hlf) * 72 + nt * 16 + ln] = f32_to_bf16(s[nt][v]);

      v16bf pf0 = load_frag_a(pw, 72, 0, 0, lane);
      v16bf pf1 = load_frag_a(pw, 72, 0, 32, lane);

      // l += rowsum(P) by WMMA against all-ones B (broadcast across columns)
      accL[mt] = wmma_bf16(pf0, onesB, accL[mt]);
      accL[mt] = wmma_bf16(pf1, onesB, accL[mt]);

      // O(16x64) += P(16x64) * V(64x64)   (V staged as [d][j])
#pragma unroll
      for (int nt = 0; nt < 4; ++nt) {
        accO[mt][nt] =
            wmma_bf16(pf0, load_frag_b_nk(sVt, 72, nt * 16, 0, lane), accO[mt][nt]);
        accO[mt][nt] =
            wmma_bf16(pf1, load_frag_b_nk(sVt, 72, nt * 16, 32, lane), accO[mt][nt]);
      }
    }
  }

  // finalize: O /= l, store bf16 merged-heads [B][SEQ][INNER]
  const size_t obase = (size_t)b * SEQ * INNER;
#pragma unroll
  for (int mt = 0; mt < 2; ++mt) {
    float linv[8];
#pragma unroll
    for (int v = 0; v < 8; ++v) linv[v] = __fdividef(1.0f, accL[mt][v]);
#pragma unroll
    for (int nt = 0; nt < 4; ++nt)
#pragma unroll
      for (int v = 0; v < 8; ++v) {
        int row = q0 + wid * 32 + mt * 16 + v + 8 * hlf;
        int col = h * DHEAD + nt * 16 + ln;
        Og[obase + (size_t)row * INNER + col] = f32_to_bf16(accO[mt][nt][v] * linv[v]);
      }
  }
}

// =====================================================================
extern "C" void kernel_launch(void* const* d_in, const int* in_sizes, int n_in,
                              void* d_out, int out_size, void* d_ws, size_t ws_size,
                              hipStream_t stream) {
  (void)in_sizes; (void)n_in; (void)out_size; (void)ws_size;
  const float* x   = (const float*)d_in[0];
  const float* ctx = (const float*)d_in[1];
  const float* wq  = (const float*)d_in[2];
  const float* bq  = (const float*)d_in[3];
  const float* wk  = (const float*)d_in[4];
  const float* bk  = (const float*)d_in[5];
  const float* wv  = (const float*)d_in[6];
  const float* bv  = (const float*)d_in[7];
  const float* wo  = (const float*)d_in[8];
  const float* bo  = (const float*)d_in[9];
  float* out = (float*)d_out;

  const size_t QE = (size_t)NBATCH * SEQ * INNER;  // elements per bf16 buffer
  unsigned short* Q = (unsigned short*)d_ws;
  unsigned short* K = Q + QE;
  unsigned short* V = K + QE;
  unsigned short* O = V + QE;   // 32 MB total bf16 workspace

  const float qscale = 0.125f;  // DIM_HEAD^-0.5 = 1/8, folded into Q

  dim3 blk256(256);
  dim3 gqkv(INNER / 64, (NBATCH * SEQ) / 128);  // (8, 64)
  proj_kernel<float, 0><<<gqkv, blk256, 0, stream>>>(x,   wq, bq, (void*)Q, DMODEL, INNER, qscale);
  proj_kernel<float, 0><<<gqkv, blk256, 0, stream>>>(ctx, wk, bk, (void*)K, DMODEL, INNER, 1.0f);
  proj_kernel<float, 0><<<gqkv, blk256, 0, stream>>>(ctx, wv, bv, (void*)V, DMODEL, INNER, 1.0f);

  dim3 blk128(128);
  dim3 gat(SEQ / 128, H_HEADS, NBATCH);  // (16, 8, 4)
  attn_kernel<<<gat, blk128, 0, stream>>>(Q, K, V, O);

  dim3 gout(DMODEL / 64, (NBATCH * SEQ) / 128);  // (16, 64)
  proj_kernel<unsigned short, 1><<<gout, blk256, 0, stream>>>(O, wo, bo, (void*)out,
                                                              INNER, DMODEL, 1.0f);
}
